// Attention_53283364274411
// MI455X (gfx1250) — matrix operand order repaired and checked by reading, not compiled, not run on hardware
//
#include <hip/hip_runtime.h>

// ---------------------------------------------------------------------------
// MI455X (gfx1250) fused attention: bf16 WMMA GEMMs + flash attention.
// Compute-bound (~0.48 TFLOP vs ~13us of pure HBM traffic) -> all matmuls via
// v_wmma_f32_16x16x32_bf16 (f32 accumulate).  128x128 double-buffered GEMM
// tiles + global_prefetch_b8 two tiles ahead; launch_bounds(...,1) keeps the
// 64-VGPR accumulators + fragments resident (no scratch spills).
// ---------------------------------------------------------------------------

constexpr int SEQ = 4096;        // NUM_SEQS * BLOCK
constexpr int DIM = 4096;
constexpr int NH  = 32;
constexpr int NKV = 8;
constexpr int HD  = 128;
constexpr int QKD = NH * HD;     // 4096
constexpr int KVD = NKV * HD;    // 1024
constexpr int BLK = 1024;

typedef __attribute__((ext_vector_type(16))) __bf16 bf16x16;
typedef __attribute__((ext_vector_type(8)))  float  f32x8;

union FragU { unsigned int u[8]; bf16x16 v; };
union Ld32  { uint4 q[2]; unsigned int w[8]; };

__device__ __forceinline__ unsigned short f2bf(float f) {
  union { float f; unsigned int u; } x; x.f = f;
  unsigned int u = x.u;
  unsigned int r = (u + 0x7FFFu + ((u >> 16) & 1u)) >> 16;
  return (unsigned short)r;
}
__device__ __forceinline__ float bf2f(unsigned short b) {
  union { unsigned int u; float f; } x; x.u = ((unsigned int)b) << 16;
  return x.f;
}

// ---------------------------------------------------------------------------
// fp32 -> bf16 bulk conversion (4 elements / thread)
// ---------------------------------------------------------------------------
__global__ void cvt_f32_bf16(const float* __restrict__ in,
                             unsigned short* __restrict__ out, int n4) {
  int i = blockIdx.x * blockDim.x + threadIdx.x;
  if (i >= n4) return;
  float4 f = ((const float4*)in)[i];
  uint2 o;
  o.x = (unsigned)f2bf(f.x) | ((unsigned)f2bf(f.y) << 16);
  o.y = (unsigned)f2bf(f.z) | ((unsigned)f2bf(f.w) << 16);
  ((uint2*)out)[i] = o;
}

// ---------------------------------------------------------------------------
// Tiled bf16 GEMM: C[M,N] = A[M,K] * B[K,N].
// 128x128 tile / 256-thread block (8 waves in a 2x4 grid; each wave a 64x32
// sub-tile = 8 accumulators), BK = 32, double-buffered LDS (ping-pong) so the
// global loads of tile k+1 overlap the 8 WMMAs of tile k, plus
// global_prefetch_b8 of tile k+2 to cover L2 latency beyond one stage.
// LDS layouts match CDNA5 WMMA fragment layouts -> fragment reads are packed
// 32-bit / b128 ds loads.
// ---------------------------------------------------------------------------
template <bool OUT_F32>
__global__ __launch_bounds__(256, 1) void gemm_bf16(
    const unsigned short* __restrict__ A, const unsigned short* __restrict__ B,
    void* __restrict__ Cv, int M, int N, int K) {
  __shared__ __align__(16) unsigned int   As[2][128 * 16];  // [row][k/2]
  __shared__ __align__(16) unsigned short Bs[2][128 * 32];  // [n][k] transp.

  const int t    = threadIdx.x;
  const int wv   = t >> 5;
  const int wm   = wv >> 2;          // 0..1 : 64-row strip
  const int wn   = wv & 3;           // 0..3 : 32-col strip
  const int lane = t & 31;
  const int half = lane >> 4;
  const int l16  = lane & 15;
  const int m0   = blockIdx.y * 128;
  const int n0   = blockIdx.x * 128;

  const int arow = t >> 1, aseg = t & 1;  // A tile: 2 thr/row, 16 bf16 each
  const int brow = t >> 3, bseg = t & 7;  // B tile: 8 thr/row, 16 bf16 each

  f32x8 acc[4][2] = {};
  Ld32 abuf, bbuf;

  // ---- preload tile 0 ----
  {
    const uint4* ga = (const uint4*)(A + (size_t)(m0 + arow) * K + aseg * 16);
    abuf.q[0] = ga[0]; abuf.q[1] = ga[1];
    const uint4* gb = (const uint4*)(B + (size_t)brow * N + n0 + bseg * 16);
    bbuf.q[0] = gb[0]; bbuf.q[1] = gb[1];
    uint4* sp = (uint4*)&As[0][arow * 16 + aseg * 8];
    sp[0] = abuf.q[0]; sp[1] = abuf.q[1];
#pragma unroll
    for (int i = 0; i < 8; ++i) {
      unsigned int w = bbuf.w[i];
      int n = bseg * 16 + 2 * i;
      Bs[0][(n)     * 32 + brow] = (unsigned short)(w & 0xFFFF);
      Bs[0][(n + 1) * 32 + brow] = (unsigned short)(w >> 16);
    }
  }
  __syncthreads();

  int cur = 0;
  for (int k0 = 0; k0 < K; k0 += 32) {
    const bool hasnext = (k0 + 32) < K;
    if (hasnext) {
      const uint4* ga =
          (const uint4*)(A + (size_t)(m0 + arow) * K + (k0 + 32) + aseg * 16);
      abuf.q[0] = ga[0]; abuf.q[1] = ga[1];
      const uint4* gb =
          (const uint4*)(B + (size_t)(k0 + 32 + brow) * N + n0 + bseg * 16);
      bbuf.q[0] = gb[0]; bbuf.q[1] = gb[1];
    }
    if (k0 + 64 < K) {  // global_prefetch_b8 two tiles ahead
      __builtin_prefetch(A + (size_t)(m0 + arow) * K + (k0 + 64) + aseg * 16,
                         0, 3);
      __builtin_prefetch(B + (size_t)(k0 + 64 + brow) * N + n0 + bseg * 16, 0,
                         3);
    }

    // ---- fragments from current buffer + 8 WMMAs ----
    FragU af[4], bfr[2];
#pragma unroll
    for (int mi = 0; mi < 4; ++mi)
#pragma unroll
      for (int r = 0; r < 8; ++r) {
        int kk = ((r & 4) ? 16 : 0) + (half ? 8 : 0) + 2 * (r & 3);
        af[mi].u[r] = As[cur][(wm * 64 + mi * 16 + l16) * 16 + (kk >> 1)];
      }
#pragma unroll
    for (int ni = 0; ni < 2; ++ni)
#pragma unroll
      for (int r = 0; r < 8; ++r) {
        int kk = ((r & 4) ? 16 : 0) + (half ? 8 : 0) + 2 * (r & 3);
        bfr[ni].u[r] = *(const unsigned int*)&Bs[cur][(wn * 32 + ni * 16 + l16) * 32 + kk];
      }
#pragma unroll
    for (int mi = 0; mi < 4; ++mi)
#pragma unroll
      for (int ni = 0; ni < 2; ++ni)
        acc[mi][ni] = __builtin_amdgcn_wmma_f32_16x16x32_bf16(
            false, af[mi].v, false, bfr[ni].v, (short)0, acc[mi][ni], false,
            false);

    // ---- stage next tile into the other buffer ----
    if (hasnext) {
      int nxt = cur ^ 1;
      uint4* sp = (uint4*)&As[nxt][arow * 16 + aseg * 8];
      sp[0] = abuf.q[0]; sp[1] = abuf.q[1];
#pragma unroll
      for (int i = 0; i < 8; ++i) {
        unsigned int w = bbuf.w[i];
        int n = bseg * 16 + 2 * i;
        Bs[nxt][(n)     * 32 + brow] = (unsigned short)(w & 0xFFFF);
        Bs[nxt][(n + 1) * 32 + brow] = (unsigned short)(w >> 16);
      }
    }
    __syncthreads();
    cur ^= 1;
  }

  // C layout: N = lane&15, M = r + 8*half
#pragma unroll
  for (int mi = 0; mi < 4; ++mi)
#pragma unroll
    for (int ni = 0; ni < 2; ++ni)
#pragma unroll
      for (int r = 0; r < 8; ++r) {
        int row = m0 + wm * 64 + mi * 16 + (half ? 8 : 0) + r;
        int col = n0 + wn * 32 + ni * 16 + l16;
        float val = acc[mi][ni][r];
        if (OUT_F32) ((float*)Cv)[(size_t)row * N + col] = val;
        else ((unsigned short*)Cv)[(size_t)row * N + col] = f2bf(val);
      }
}

// ---------------------------------------------------------------------------
// RoPE, interleaved-pair convention, in place on bf16 tensor (SEQ, nh, 128).
// One packed 32-bit load/store per rotation pair.
// ---------------------------------------------------------------------------
__global__ void rope_kernel(unsigned short* __restrict__ q,
                            const float* __restrict__ cosb,
                            const float* __restrict__ sinb, int nheads) {
  int idx = blockIdx.x * blockDim.x + threadIdx.x;   // SEQ*nheads*64 threads
  int i = idx & 63;
  int h = (idx >> 6) % nheads;
  int s = idx / (64 * nheads);
  if (s >= SEQ) return;
  unsigned int* p =
      (unsigned int*)(q + ((size_t)s * nheads + h) * HD + 2 * i);
  float c  = cosb[s * 64 + i];
  float sn = sinb[s * 64 + i];
  unsigned int w = *p;
  float t0 = bf2f((unsigned short)(w & 0xFFFF));
  float t1 = bf2f((unsigned short)(w >> 16));
  unsigned int r0 = f2bf(t0 * c - t1 * sn);
  unsigned int r1 = f2bf(t0 * sn + t1 * c);
  *p = r0 | (r1 << 16);
}

// ---------------------------------------------------------------------------
// Flash attention (causal, GQA 4:1).  grid = (16 q-tiles, 4 blocks * 32 heads)
// 128 threads = 4 waves; each wave owns 16 query rows.  K chunk (32x128) and
// V chunk (transposed) staged in LDS and shared by all waves; P transposed
// C-layout -> A-layout through a per-wave LDS tile; PV uses K=32 WMMA.
// launch_bounds(128,1): keep Q frags (32 VGPRs) + O accum (64 VGPRs) resident.
// ---------------------------------------------------------------------------
__global__ __launch_bounds__(128, 1) void flash_attn(
    const unsigned short* __restrict__ Q, const unsigned short* __restrict__ Kc,
    const unsigned short* __restrict__ V, unsigned short* __restrict__ O) {
  __shared__ __align__(16) unsigned int Ks[32 * 64];   // [key][d/2]   8 KB
  __shared__ __align__(16) unsigned int Vt[128 * 16];  // [d][key/2]   8 KB
  __shared__ __align__(16) unsigned int Ps[4][256];    // per-wave 16x32 bf16

  const int t    = threadIdx.x;
  const int wv   = t >> 5;
  const int lane = t & 31;
  const int half = lane >> 4;
  const int l16  = lane & 15;

  const int bh  = blockIdx.y;
  const int b   = bh >> 5;
  const int h   = bh & 31;
  const int hkv = h >> 2;
  const int qt  = blockIdx.x * 64 + wv * 16;   // query base within seq block

  // Preload Q fragments for the 4 K=32 steps covering D=128 (reused all loop)
  FragU aq[4];
#pragma unroll
  for (int s = 0; s < 4; ++s) {
#pragma unroll
    for (int r = 0; r < 8; ++r) {
      int d = s * 32 + ((r & 4) ? 16 : 0) + (half ? 8 : 0) + 2 * (r & 3);
      const unsigned int* qp = (const unsigned int*)(
          Q + (size_t)(b * BLK + qt + l16) * QKD + h * HD + d);
      aq[s].u[r] = *qp;
    }
  }

  float rm[8], rl[8];
#pragma unroll
  for (int r = 0; r < 8; ++r) { rm[r] = -3.0e38f; rl[r] = 0.0f; }
  f32x8 o[8] = {};

  const float scale = 0.08838834764831845f;  // 1/sqrt(128)
  const int nch = blockIdx.x * 2 + 2;        // key chunks of 32 (causal bound)

  for (int ch = 0; ch < nch; ++ch) {
    const int kc = ch * 32;
    // ---- cooperative load: K chunk [32 keys][128 d] ----
    {
      int row = t >> 2, seg = t & 3;
      const uint4* gp = (const uint4*)(
          Kc + (size_t)(b * BLK + kc + row) * KVD + hkv * HD + seg * 32);
      uint4* sp = (uint4*)&Ks[row * 64 + seg * 16];
#pragma unroll
      for (int i = 0; i < 4; ++i) sp[i] = gp[i];
    }
    // ---- cooperative load: V chunk transposed -> Vt[d][key] ----
    {
      int row = t >> 2, seg = t & 3;
      const unsigned int* gp = (const unsigned int*)(
          V + (size_t)(b * BLK + kc + row) * KVD + hkv * HD + seg * 32);
      unsigned short* vp = (unsigned short*)Vt;
#pragma unroll
      for (int i = 0; i < 16; ++i) {
        unsigned int w = gp[i];
        int d = seg * 32 + 2 * i;
        vp[(size_t)d * 32 + row]       = (unsigned short)(w & 0xFFFF);
        vp[(size_t)(d + 1) * 32 + row] = (unsigned short)(w >> 16);
      }
    }
    if (ch + 1 < nch) {  // prefetch next chunk's K/V rows into L2/L0
      int row = t >> 2, seg = t & 3;
      __builtin_prefetch(
          Kc + (size_t)(b * BLK + kc + 32 + row) * KVD + hkv * HD + seg * 32,
          0, 3);
      __builtin_prefetch(
          V + (size_t)(b * BLK + kc + 32 + row) * KVD + hkv * HD + seg * 32, 0,
          3);
    }
    __syncthreads();

    // ---- scores for the two 16-key sub-tiles ----
    f32x8 sc[2];
#pragma unroll
    for (int kt = 0; kt < 2; ++kt) {
      f32x8 s0 = {};
#pragma unroll
      for (int s = 0; s < 4; ++s) {
        FragU bf_;
#pragma unroll
        for (int r = 0; r < 8; ++r) {
          int d = s * 32 + ((r & 4) ? 16 : 0) + (half ? 8 : 0) + 2 * (r & 3);
          bf_.u[r] = Ks[(kt * 16 + l16) * 64 + (d >> 1)];
        }
        s0 = __builtin_amdgcn_wmma_f32_16x16x32_bf16(
            false, aq[s].v, false, bf_.v, (short)0, s0, false, false);
      }
      int keyn = kc + kt * 16 + l16;
#pragma unroll
      for (int r = 0; r < 8; ++r) {
        int qrow = qt + (half ? 8 : 0) + r;
        float v = s0[r] * scale;
        s0[r] = (keyn <= qrow) ? v : -3.0e38f;
      }
      sc[kt] = s0;
    }

    // ---- online softmax (row = half-wave of 16 lanes, one VGPR per row) ----
    float corr[8];
    unsigned short* psw = (unsigned short*)&Ps[wv][0];
#pragma unroll
    for (int r = 0; r < 8; ++r) {
      float mx = fmaxf(sc[0][r], sc[1][r]);
      mx = fmaxf(mx, __shfl_xor(mx, 1, 16));
      mx = fmaxf(mx, __shfl_xor(mx, 2, 16));
      mx = fmaxf(mx, __shfl_xor(mx, 4, 16));
      mx = fmaxf(mx, __shfl_xor(mx, 8, 16));
      float mnew = fmaxf(rm[r], mx);
      corr[r] = __expf(rm[r] - mnew);
      rm[r] = mnew;
      float p0 = __expf(sc[0][r] - mnew);
      float p1 = __expf(sc[1][r] - mnew);
      float rs = p0 + p1;
      rs += __shfl_xor(rs, 1, 16);
      rs += __shfl_xor(rs, 2, 16);
      rs += __shfl_xor(rs, 4, 16);
      rs += __shfl_xor(rs, 8, 16);
      rl[r] = rl[r] * corr[r] + rs;
      int qr = (half ? 8 : 0) + r;       // C-layout row
      psw[qr * 32 + l16]      = f2bf(p0);
      psw[qr * 32 + 16 + l16] = f2bf(p1);
    }
#pragma unroll
    for (int j = 0; j < 8; ++j)
#pragma unroll
      for (int r = 0; r < 8; ++r) o[j][r] *= corr[r];

    // ---- reload P as an A-fragment (16 q x 32 keys) ----
    FragU pf;
#pragma unroll
    for (int r = 0; r < 8; ++r) {
      int kk = ((r & 4) ? 16 : 0) + (half ? 8 : 0) + 2 * (r & 3);
      pf.u[r] = *(const unsigned int*)(psw + l16 * 32 + kk);
    }
    // ---- PV: accumulate 16x128 output (8 WMMA tiles, K = 32 keys) ----
#pragma unroll
    for (int j = 0; j < 8; ++j) {
      FragU bv;
#pragma unroll
      for (int r = 0; r < 8; ++r) {
        int kk = ((r & 4) ? 16 : 0) + (half ? 8 : 0) + 2 * (r & 3);
        bv.u[r] = Vt[(j * 16 + l16) * 16 + (kk >> 1)];
      }
      o[j] = __builtin_amdgcn_wmma_f32_16x16x32_bf16(
          false, pf.v, false, bv.v, (short)0, o[j], false, false);
    }
    __syncthreads();
  }

  // ---- normalize and emit bf16 attention output (S, H*D) ----
#pragma unroll
  for (int r = 0; r < 8; ++r) {
    float inv = 1.0f / rl[r];
    int qrow = b * BLK + qt + (half ? 8 : 0) + r;
#pragma unroll
    for (int j = 0; j < 8; ++j) {
      O[(size_t)qrow * QKD + h * HD + j * 16 + l16] = f2bf(o[j][r] * inv);
    }
  }
}

// ---------------------------------------------------------------------------
// Host-side orchestration
// ---------------------------------------------------------------------------
extern "C" void kernel_launch(void* const* d_in, const int* in_sizes, int n_in,
                              void* d_out, int out_size, void* d_ws,
                              size_t ws_size, hipStream_t stream) {
  const float* x    = (const float*)d_in[0];
  const float* wq   = (const float*)d_in[1];
  const float* wk   = (const float*)d_in[2];
  const float* wv   = (const float*)d_in[3];
  const float* wo   = (const float*)d_in[4];
  const float* fcos = (const float*)d_in[5];
  const float* fsin = (const float*)d_in[6];

  char* ws = (char*)d_ws;
  size_t off = 0;
  auto take = [&](size_t bytes) -> char* {
    char* p = ws + off;
    off = (off + bytes + 255) & ~(size_t)255;
    return p;
  };
  unsigned short* xb  = (unsigned short*)take((size_t)SEQ * DIM * 2);
  unsigned short* wqb = (unsigned short*)take((size_t)DIM * QKD * 2);
  unsigned short* wkb = (unsigned short*)take((size_t)DIM * KVD * 2);
  unsigned short* wvb = (unsigned short*)take((size_t)DIM * KVD * 2);
  unsigned short* wob = (unsigned short*)take((size_t)QKD * DIM * 2);
  unsigned short* qb  = (unsigned short*)take((size_t)SEQ * QKD * 2);
  unsigned short* kb  = (unsigned short*)take((size_t)SEQ * KVD * 2);
  unsigned short* vb  = (unsigned short*)take((size_t)SEQ * KVD * 2);
  unsigned short* ob  = (unsigned short*)take((size_t)SEQ * QKD * 2);

  auto cvt = [&](const float* src, unsigned short* dst, size_t n) {
    int n4 = (int)(n / 4);
    cvt_f32_bf16<<<(n4 + 255) / 256, 256, 0, stream>>>(src, dst, n4);
  };
  cvt(x,  xb,  (size_t)SEQ * DIM);
  cvt(wq, wqb, (size_t)DIM * QKD);
  cvt(wk, wkb, (size_t)DIM * KVD);
  cvt(wv, wvb, (size_t)DIM * KVD);
  cvt(wo, wob, (size_t)QKD * DIM);

  // QKV projections (bf16 out), 128x128 tiles
  gemm_bf16<false><<<dim3(QKD / 128, SEQ / 128), 256, 0, stream>>>(
      xb, wqb, qb, SEQ, QKD, DIM);
  gemm_bf16<false><<<dim3(KVD / 128, SEQ / 128), 256, 0, stream>>>(
      xb, wkb, kb, SEQ, KVD, DIM);
  gemm_bf16<false><<<dim3(KVD / 128, SEQ / 128), 256, 0, stream>>>(
      xb, wvb, vb, SEQ, KVD, DIM);

  // RoPE on Q and K
  rope_kernel<<<(SEQ * NH * 64) / 256, 256, 0, stream>>>(qb, fcos, fsin, NH);
  rope_kernel<<<(SEQ * NKV * 64) / 256, 256, 0, stream>>>(kb, fcos, fsin, NKV);

  // Causal flash attention per (q-tile, block*head)
  flash_attn<<<dim3(BLK / 64, (SEQ / BLK) * NH), 128, 0, stream>>>(qb, kb, vb,
                                                                   ob);

  // Output projection (fp32 out -> d_out)
  gemm_bf16<true><<<dim3(DIM / 128, SEQ / 128), 256, 0, stream>>>(
      ob, wob, (float*)d_out, SEQ, DIM, QKD);
}
